// SelfAttention_10213432230532
// MI455X (gfx1250) — matrix operand order repaired and checked by reading, not compiled, not run on hardware
//
#include <hip/hip_runtime.h>

typedef __bf16 bf16;
typedef __attribute__((ext_vector_type(16))) __bf16 v16bf;
typedef __attribute__((ext_vector_type(8)))  float  v8f;

union FragU { uint4 u[2]; v16bf v; };

// A-frag (16x32 bf16, row-major source, stride in elements).
// lane&15 = row; lane>>4 = half. Runs: [half*8 .. +7] and [16+half*8 .. +7].
__device__ inline v16bf load_frag_A(const bf16* base, int stride, int lane) {
    int row = lane & 15, half = lane >> 4;
    const bf16* p = base + (size_t)row * stride + half * 8;
    FragU f;
    f.u[0] = *(const uint4*)p;
    f.u[1] = *(const uint4*)(p + 16);
    return f.v;
}

// B-frag (32x16 bf16) from K-contiguous storage Bt[n][k] (stride in elements).
// lane&15 = col n; lane>>4 = half. Runs: K = [half*8 .. +7] and [16+half*8 .. +7].
__device__ inline v16bf load_frag_B(const bf16* base, int stride, int lane) {
    int n = lane & 15, half = lane >> 4;
    const bf16* p = base + (size_t)n * stride + half * 8;
    FragU f;
    f.u[0] = *(const uint4*)p;
    f.u[1] = *(const uint4*)(p + 16);
    return f.v;
}

__device__ inline v8f wmma_bf16(v16bf a, v16bf b, v8f c) {
    return __builtin_amdgcn_wmma_f32_16x16x32_bf16(
        false, a, false, b, (short)0, c, false, false);
}

// ISA 10.2: for generic pointers into LDS, addr[31:0] IS the LDS byte address.
__device__ inline unsigned lds_off(const void* p) {
    return (unsigned)(uintptr_t)p;
}

// Async copy-engine staging: 16B global -> LDS, tracked by ASYNCcnt.
__device__ inline void async_b128(unsigned lds, const bf16* sbase, unsigned voff) {
    asm volatile("global_load_async_to_lds_b128 %0, %1, %2"
                 :: "v"(lds), "v"(voff),
                    "s"((unsigned long long)(uintptr_t)sbase)
                 : "memory");
}
// Async loads complete in order: waiting <=4 means "current tile landed,
// next tile's 4 copies may still be in flight".
__device__ inline void wait_async_le4() {
    asm volatile("s_wait_asynccnt 0x4" ::: "memory");
}
__device__ inline void wait_async_le0() {
    asm volatile("s_wait_asynccnt 0x0" ::: "memory");
}

#define BM 128
#define BN 128
#define BK 32

__device__ inline void stage_tile(const bf16* A, const bf16* BT, int Kd,
                                  int m0, int n0, int kt,
                                  unsigned ldsA, unsigned ldsB,
                                  int row, int colb) {
    unsigned aoff = (unsigned)(((size_t)(m0 + row) * Kd + kt + colb) * 2);
    unsigned boff = (unsigned)(((size_t)(n0 + row) * Kd + kt + colb) * 2);
    async_b128(ldsA,      A,  aoff);
    async_b128(ldsA + 16, A,  aoff + 16);
    async_b128(ldsB,      BT, boff);
    async_b128(ldsB + 16, BT, boff + 16);
}

// GEMM mainloop: C[BMxBN] += A[BMxK] * BT[NxK]^T, bf16, double-buffered LDS,
// async copies for tile i+1 overlap the 8 WMMAs of tile i.
// 8 waves: 2 (M) x 4 (N); each wave 64x32 = 4x2 fragments.
__device__ inline void gemm_tiles(const bf16* __restrict__ A,
                                  const bf16* __restrict__ BT,
                                  int Kd, int m0, int n0,
                                  bf16 (*At)[BM][BK], bf16 (*Bt)[BM][BK],
                                  v8f acc[4][2]) {
    int tid = threadIdx.x, lane = tid & 31, wave = tid >> 5;
    int waveM = (wave >> 2) * 64;
    int waveN = (wave & 3) * 32;
    int row = tid >> 1, colb = (tid & 1) * 16;     // thread -> 32B half-row
    unsigned ldsA[2] = { lds_off(&At[0][row][colb]), lds_off(&At[1][row][colb]) };
    unsigned ldsB[2] = { lds_off(&Bt[0][row][colb]), lds_off(&Bt[1][row][colb]) };

    stage_tile(A, BT, Kd, m0, n0, 0, ldsA[0], ldsB[0], row, colb);
    int nk = Kd / BK;
    for (int it = 0; it < nk; ++it) {
        int cur = it & 1;
        __syncthreads();     // all waves done reading buf[cur^1] (prev compute)
        if (it + 1 < nk) {
            stage_tile(A, BT, Kd, m0, n0, (it + 1) * BK,
                       ldsA[cur ^ 1], ldsB[cur ^ 1], row, colb);
            wait_async_le4();      // tile `it` landed; tile `it+1` in flight
        } else {
            wait_async_le0();
        }
        __syncthreads();     // tile `it` visible block-wide
        v16bf af[4], bfr[2];
#pragma unroll
        for (int mf = 0; mf < 4; ++mf)
            af[mf] = load_frag_A(&At[cur][waveM + mf * 16][0], BK, lane);
#pragma unroll
        for (int nf = 0; nf < 2; ++nf)
            bfr[nf] = load_frag_B(&Bt[cur][waveN + nf * 16][0], BK, lane);
#pragma unroll
        for (int mf = 0; mf < 4; ++mf)
#pragma unroll
            for (int nf = 0; nf < 2; ++nf)
                acc[mf][nf] = wmma_bf16(af[mf], bfr[nf], acc[mf][nf]);
    }
}

__global__ void cvt_f32_bf16(const float* __restrict__ src, bf16* __restrict__ dst, int n) {
    int i = blockIdx.x * blockDim.x + threadIdx.x;
    int stride = gridDim.x * blockDim.x;
    for (; i < n; i += stride) dst[i] = (bf16)src[i];
}

// One-time weight transpose+convert: src[K][N] f32 -> dst[N][K] bf16.
__global__ void __launch_bounds__(256)
transpose_cvt(const float* __restrict__ src, bf16* __restrict__ dst, int K, int N) {
    __shared__ float t[32][33];
    int n0 = blockIdx.x * 32, k0 = blockIdx.y * 32;
    int tx = threadIdx.x, ty = threadIdx.y;        // 32 x 8
#pragma unroll
    for (int i = 0; i < 32; i += 8)
        t[ty + i][tx] = src[(size_t)(k0 + ty + i) * N + n0 + tx];
    __syncthreads();
#pragma unroll
    for (int i = 0; i < 32; i += 8)
        dst[(size_t)(n0 + ty + i) * K + k0 + tx] = (bf16)t[tx][ty + i];
}

// QKV projection: x[8192,1024] @ w_in -> q/k [B,H,L,64] bf16, v stored
// TRANSPOSED as [B,H,64,L] bf16 so attention needs no LDS transpose.
// q gets the 1/sqrt(64)=0.125 logit scale folded in.
__global__ void __launch_bounds__(256)
qkv_gemm(const bf16* __restrict__ xb, const bf16* __restrict__ wT,
         const float* __restrict__ b_in,
         bf16* __restrict__ qh, bf16* __restrict__ kh, bf16* __restrict__ vt) {
    __shared__ __align__(16) bf16 At[2][BM][BK];
    __shared__ __align__(16) bf16 Bt[2][BM][BK];
    int n0 = blockIdx.x * BN, m0 = blockIdx.y * BM;
    v8f acc[4][2] = {};
    gemm_tiles(xb, wT, 1024, m0, n0, At, Bt, acc);
    int lane = threadIdx.x & 31, wave = threadIdx.x >> 5;
    int waveM = (wave >> 2) * 64, waveN = (wave & 3) * 32;
#pragma unroll
    for (int mf = 0; mf < 4; ++mf)
#pragma unroll
        for (int nf = 0; nf < 2; ++nf) {
            int mbase = m0 + waveM + mf * 16 + ((lane & 16) ? 8 : 0);
            int n = n0 + waveN + nf * 16 + (lane & 15);
            float bias = b_in[n];
            int sel = n >> 10, d = n & 1023, h = d >> 6, kk = d & 63;
#pragma unroll
            for (int r = 0; r < 8; ++r) {
                int m = mbase + r, b = m >> 10, l = m & 1023;
                size_t bh = (size_t)(b * 16 + h);
                float val = acc[mf][nf][r] + bias;
                if (sel == 0)
                    qh[((bh << 10) + l) * 64 + kk] = (bf16)(val * 0.125f);
                else if (sel == 1)
                    kh[((bh << 10) + l) * 64 + kk] = (bf16)val;
                else
                    vt[((bh * 64 + kk) << 10) + l] = (bf16)val;   // transposed
            }
        }
}

// Fused (q k^T) v per (b,h); no softmax (reference reproduces that bug).
// Barrier-free: every wave owns 16 q rows x all 64 out cols. k is B-frag
// contiguous (B=k^T), vT is B-frag contiguous; only the S f32->A-frag bf16
// re-layout bounces through per-wave LDS (same-wave DS ordering).
__global__ void __launch_bounds__(256)
attn_kernel(const bf16* __restrict__ qh, const bf16* __restrict__ kh,
            const bf16* __restrict__ vt, bf16* __restrict__ ao) {
    __shared__ __align__(16) bf16 Sb[8][16][32];
    int bh = blockIdx.y;
    int b = bh >> 4, h = bh & 15;
    size_t base = (size_t)bh * 1024 * 64;
    const bf16* qp = qh + base;
    const bf16* kp = kh + base;
    const bf16* vp = vt + base;                  // [64][1024]
    int tid = threadIdx.x, lane = tid & 31, wave = tid >> 5;
    int mrow = blockIdx.x * 128 + wave * 16;

    v16bf qf[2];
#pragma unroll
    for (int ks = 0; ks < 2; ++ks)
        qf[ks] = load_frag_A(qp + (size_t)mrow * 64 + ks * 32, 64, lane);

    v8f oacc[4] = {};
    for (int sc = 0; sc < 32; ++sc) {
        int sbase = sc * 32;
        // S = q @ k^T chunk (16x32).
        v8f sacc[2] = {};
#pragma unroll
        for (int nf = 0; nf < 2; ++nf)
#pragma unroll
            for (int ks = 0; ks < 2; ++ks) {
                v16bf kf = load_frag_B(kp + (size_t)(sbase + nf * 16) * 64 + ks * 32,
                                       64, lane);
                sacc[nf] = wmma_bf16(qf[ks], kf, sacc[nf]);
            }
        // C-layout f32 -> A-frag bf16 via per-wave LDS bounce.
#pragma unroll
        for (int nf = 0; nf < 2; ++nf)
#pragma unroll
            for (int r = 0; r < 8; ++r) {
                int row = ((lane & 16) ? 8 : 0) + r;
                Sb[wave][row][nf * 16 + (lane & 15)] = (bf16)sacc[nf][r];
            }
        v16bf sf = load_frag_A(&Sb[wave][0][0], 32, lane);
        // O += S @ v_chunk; vT rows are kk, contiguous in s.
#pragma unroll
        for (int nf = 0; nf < 4; ++nf) {
            v16bf vf = load_frag_B(vp + (size_t)(nf * 16) * 1024 + sbase, 1024, lane);
            oacc[nf] = wmma_bf16(sf, vf, oacc[nf]);
        }
    }
    // O -> attn_out[b, l, h*64+kk] bf16
#pragma unroll
    for (int nf = 0; nf < 4; ++nf) {
        int col = h * 64 + nf * 16 + (lane & 15);
#pragma unroll
        for (int r = 0; r < 8; ++r) {
            int l = mrow + ((lane & 16) ? 8 : 0) + r;
            ao[((size_t)b * 1024 + l) * 1024 + col] = (bf16)oacc[nf][r];
        }
    }
}

// Output projection: [8192,1024] @ w_out + b_out -> f32 d_out.
__global__ void __launch_bounds__(256)
out_gemm(const bf16* __restrict__ ab, const bf16* __restrict__ wT,
         const float* __restrict__ b_out, float* __restrict__ out) {
    __shared__ __align__(16) bf16 At[2][BM][BK];
    __shared__ __align__(16) bf16 Bt[2][BM][BK];
    int n0 = blockIdx.x * BN, m0 = blockIdx.y * BM;
    v8f acc[4][2] = {};
    gemm_tiles(ab, wT, 1024, m0, n0, At, Bt, acc);
    int lane = threadIdx.x & 31, wave = threadIdx.x >> 5;
    int waveM = (wave >> 2) * 64, waveN = (wave & 3) * 32;
#pragma unroll
    for (int mf = 0; mf < 4; ++mf)
#pragma unroll
        for (int nf = 0; nf < 2; ++nf) {
            int mbase = m0 + waveM + mf * 16 + ((lane & 16) ? 8 : 0);
            int n = n0 + waveN + nf * 16 + (lane & 15);
            float bias = b_out[n];
#pragma unroll
            for (int r = 0; r < 8; ++r)
                out[(size_t)(mbase + r) * 1024 + n] = acc[mf][nf][r] + bias;
        }
}

extern "C" void kernel_launch(void* const* d_in, const int* in_sizes, int n_in,
                              void* d_out, int out_size, void* d_ws, size_t ws_size,
                              hipStream_t stream) {
    (void)in_sizes; (void)n_in; (void)out_size; (void)ws_size;
    const float* x     = (const float*)d_in[0];
    const float* w_in  = (const float*)d_in[1];
    const float* b_in  = (const float*)d_in[2];
    const float* w_out = (const float*)d_in[3];
    const float* b_out = (const float*)d_in[4];
    float* out = (float*)d_out;

    const int NX  = 8192 * 1024;
    const int NWI = 1024 * 3072;
    const int NWO = 1024 * 1024;
    const int NH  = 8 * 16 * 1024 * 64;

    bf16* xb   = (bf16*)d_ws;
    bf16* wiT  = xb  + NX;     // [3072][1024]
    bf16* woT  = wiT + NWI;    // [1024][1024]
    bf16* qh   = woT + NWO;    // [B,H,L,64]
    bf16* kh   = qh  + NH;     // [B,H,L,64]
    bf16* vt   = kh  + NH;     // [B,H,64,L]  (transposed)
    bf16* ao   = vt  + NH;     // [8192,1024]

    cvt_f32_bf16<<<2048, 256, 0, stream>>>(x, xb, NX);
    transpose_cvt<<<dim3(3072 / 32, 1024 / 32), dim3(32, 8), 0, stream>>>(w_in,  wiT, 1024, 3072);
    transpose_cvt<<<dim3(1024 / 32, 1024 / 32), dim3(32, 8), 0, stream>>>(w_out, woT, 1024, 1024);

    qkv_gemm<<<dim3(3072 / BN, 8192 / BM), 256, 0, stream>>>(xb, wiT, b_in, qh, kh, vt);
    attn_kernel<<<dim3(8, 128), 256, 0, stream>>>(qh, kh, vt, ao);
    out_gemm<<<dim3(1024 / BN, 8192 / BM), 256, 0, stream>>>(ao, woT, b_out, out);
}
